// SwinTransformerBlock_25718264168753
// MI455X (gfx1250) — compile-verified, matrix operand order checked by reading, not run on hardware
//
#include <hip/hip_runtime.h>
#include <hip/hip_bf16.h>

// ---------------------------------------------------------------------------
// Swin Transformer block for MI455X (gfx1250, wave32, WMMA f16 16x16x32).
// ---------------------------------------------------------------------------

typedef _Float16 h8  __attribute__((ext_vector_type(8)));
typedef _Float16 h16 __attribute__((ext_vector_type(16)));
typedef float    f8  __attribute__((ext_vector_type(8)));

#define DEV static __device__ __forceinline__

namespace {
constexpr int kC   = 192;     // channels
constexpr int kNH  = 6;       // heads
constexpr int kHD  = 32;      // head dim
constexpr int kN   = 49;      // tokens per 7x7 window
constexpr int kB   = 32;      // batch
constexpr int kL   = 3136;    // 56*56
constexpr int kBW  = 2048;    // total windows (32 * 64)
constexpr int kTOK = 100352;  // kBW * kN == kB * kL
constexpr int kHID = 768;
}

#define WMMA_F16(a, b, c) \
  __builtin_amdgcn_wmma_f32_16x16x32_f16(false, (a), false, (b), (short)0, (c), false, false)

// Gather one 16x32 f16 A/B^T fragment from a row-major (rows x stride) array.
// Lane l holds row (passed in); lanes 0-15 take K {k0..k0+7, k0+16..k0+23},
// lanes 16-31 take K {k0+8..k0+15, k0+24..k0+31}  (CDNA5 16-bit operand layout).
DEV h16 frag_ld(const _Float16* __restrict__ base, int stride, int row, int k0, int lane) {
  const _Float16* p = base + (size_t)row * (size_t)stride + k0 + ((lane & 16) ? 8 : 0);
  h16 r;
  ((h8*)&r)[0] = *(const h8*)(p);
  ((h8*)&r)[1] = *(const h8*)(p + 16);
  return r;
}

DEV float wave_sum32(float v) {
  for (int m = 1; m < 32; m <<= 1) v += __shfl_xor(v, m, 32);
  return v;
}
DEV float row_max16(float v) {
  for (int m = 1; m < 16; m <<= 1) v = fmaxf(v, __shfl_xor(v, m, 32));
  return v;
}
DEV float row_sum16(float v) {
  for (int m = 1; m < 16; m <<= 1) v += __shfl_xor(v, m, 32);
  return v;
}

// ---------------------------------------------------------------------------
// f32 -> f16 conversion (weights)
__global__ void k_f2h(const float* __restrict__ s, _Float16* __restrict__ d, int n) {
  int i = blockIdx.x * 256 + threadIdx.x;
  if (i < n) d[i] = (_Float16)s[i];
}

// ---------------------------------------------------------------------------
// LN1 + cyclic shift(-3,-3) + 7x7 window partition -> f16 (kBW, 49, 192)
__global__ void __launch_bounds__(256)
k_ln1_win(const float* __restrict__ x, const float* __restrict__ g,
          const float* __restrict__ bb, _Float16* __restrict__ winA) {
  int wave = threadIdx.x >> 5, lane = threadIdx.x & 31;
  int t = blockIdx.x * 8 + wave;
  if (t >= kTOK) return;
  int bw = t / kN, m = t - bw * kN;
  int b = bw >> 6, win = bw & 63;
  int wh = win >> 3, ww = win & 7;
  int i = m / 7, j = m - i * 7;
  int h = wh * 7 + i + 3; if (h >= 56) h -= 56;
  int w = ww * 7 + j + 3; if (w >= 56) w -= 56;
  const float* xp = x + ((size_t)b * kL + h * 56 + w) * kC;
  float v[6], s = 0.f;
  for (int k = 0; k < 6; ++k) { v[k] = xp[lane + 32 * k]; s += v[k]; }
  s = wave_sum32(s);
  float mean = s * (1.f / 192.f);
  float vs = 0.f;
  for (int k = 0; k < 6; ++k) { float d = v[k] - mean; vs += d * d; }
  vs = wave_sum32(vs);
  float rstd = rsqrtf(vs * (1.f / 192.f) + 1e-5f);
  _Float16* op = winA + (size_t)t * kC;
  for (int k = 0; k < 6; ++k) {
    int c = lane + 32 * k;
    op[c] = (_Float16)((v[k] - mean) * rstd * g[c] + bb[c]);
  }
}

// ---------------------------------------------------------------------------
// LN2 (no permute) -> f16 (kTOK, 192)
__global__ void __launch_bounds__(256)
k_ln2(const float* __restrict__ x, const float* __restrict__ g,
      const float* __restrict__ bb, _Float16* __restrict__ dst) {
  int wave = threadIdx.x >> 5, lane = threadIdx.x & 31;
  int t = blockIdx.x * 8 + wave;
  if (t >= kTOK) return;
  const float* xp = x + (size_t)t * kC;
  float v[6], s = 0.f;
  for (int k = 0; k < 6; ++k) { v[k] = xp[lane + 32 * k]; s += v[k]; }
  s = wave_sum32(s);
  float mean = s * (1.f / 192.f);
  float vs = 0.f;
  for (int k = 0; k < 6; ++k) { float d = v[k] - mean; vs += d * d; }
  vs = wave_sum32(vs);
  float rstd = rsqrtf(vs * (1.f / 192.f) + 1e-5f);
  _Float16* op = dst + (size_t)t * kC;
  for (int k = 0; k < 6; ++k) {
    int c = lane + 32 * k;
    op[c] = (_Float16)((v[k] - mean) * rstd * g[c] + bb[c]);
  }
}

// ---------------------------------------------------------------------------
// QKV GEMM: per window (49x192) @ (192x576) + bias -> q/k/v (kBW, 6, 49, 32) f16
__global__ void __launch_bounds__(256)
k_qkv(const _Float16* __restrict__ winA, const _Float16* __restrict__ wq,
      const float* __restrict__ qb, _Float16* __restrict__ qo,
      _Float16* __restrict__ ko, _Float16* __restrict__ vo) {
  int wave = threadIdx.x >> 5, lane = threadIdx.x & 31;
  int bw = blockIdx.x;
  int hi8 = (lane & 16) ? 8 : 0;
  const _Float16* A = winA + (size_t)bw * kN * kC;
  for (int t = wave; t < 144; t += 8) {           // 4 mtiles x 36 ntiles
    int mt = t / 36, nt = t - mt * 36;
    f8 acc = {};
    for (int kk = 0; kk < 6; ++kk) {
      h16 a = frag_ld(A, kC, mt * 16 + (lane & 15), kk * 32, lane);
      h16 b = frag_ld(wq, kC, nt * 16 + (lane & 15), kk * 32, lane);
      acc = WMMA_F16(a, b, acc);
    }
    int col = nt * 16 + (lane & 15);
    float bias = qb[col];
    int which = col / kC, rem = col - which * kC;
    int head = rem >> 5, d = rem & 31;
    _Float16* op = (which == 0 ? qo : which == 1 ? ko : vo) +
                   ((size_t)bw * kNH + head) * (size_t)(kN * kHD) + d;
    int rb = mt * 16 + hi8;
    for (int r = 0; r < 8; ++r) {
      int mrow = rb + r;
      if (mrow < kN) op[(size_t)mrow * kHD] = (_Float16)(acc[r] + bias);
    }
  }
}

// ---------------------------------------------------------------------------
// Windowed attention, one wave per (window, head).
// S = QK^T*scale + rel-pos bias + shift mask -> softmax -> P@V
__global__ void __launch_bounds__(128)
k_attn(const _Float16* __restrict__ qb_, const _Float16* __restrict__ kb_,
       const _Float16* __restrict__ vb_, const float* __restrict__ rpb,
       _Float16* __restrict__ out) {
  __shared__ _Float16 smem[4 * 6144];             // per wave: vT 32x64 + P 64x64
  int wave = threadIdx.x >> 5, lane = threadIdx.x & 31;
  int gw = blockIdx.x * 4 + wave;                 // 0 .. 12287
  int bw = gw / kNH, head = gw - bw * kNH;
  int win = bw & 63, wh = win >> 3, ww = win & 7;
  size_t off = ((size_t)bw * kNH + head) * (size_t)(kN * kHD);
  const _Float16* q = qb_ + off;
  const _Float16* k = kb_ + off;
  const _Float16* v = vb_ + off;
  _Float16* vT = smem + wave * 6144;
  _Float16* Pm = vT + 2048;
  int nloc = lane & 15;
  int hi8 = (lane & 16) ? 8 : 0;

  // stage V^T (hd x tokens), zero-pad tokens 49..63
  for (int tt = 0; tt < 64; ++tt)
    vT[lane * 64 + tt] = (tt < kN) ? v[(size_t)tt * kHD + lane] : (_Float16)0;

  // S = Q K^T  (64x64 padded, K = 32 single step)
  f8 S[4][4];
  h16 af[4];
  for (int mt = 0; mt < 4; ++mt)
    af[mt] = frag_ld(q, kHD, mt * 16 + nloc, 0, lane);
  for (int nt = 0; nt < 4; ++nt) {
    h16 bf = frag_ld(k, kHD, nt * 16 + nloc, 0, lane);
    for (int mt = 0; mt < 4; ++mt) {
      f8 z = {};
      S[mt][nt] = WMMA_F16(af[mt], bf, z);
    }
  }

  // epilogue: scale + bias + mask, rowwise softmax, write P (f16) to LDS
  const float scale = 0.1767766952966369f;        // 32^-0.5
  for (int mt = 0; mt < 4; ++mt) {
    for (int r = 0; r < 8; ++r) {
      int m = mt * 16 + hi8 + r;                  // query index (may be >= 49)
      int ia = m / 7, ja = m - ia * 7;
      int hsa = wh * 7 + ia, wsa = ww * 7 + ja;
      int rga = (hsa < 49 ? 0 : (hsa < 53 ? 1 : 2)) * 3 +
                (wsa < 49 ? 0 : (wsa < 53 ? 1 : 2));
      float vals[4];
      for (int nt = 0; nt < 4; ++nt) {
        int n = nt * 16 + nloc;                   // key index
        float sv = -1e30f;
        if (m < kN && n < kN) {
          int ib = n / 7, jb = n - ib * 7;
          int hsb = wh * 7 + ib, wsb = ww * 7 + jb;
          int rgb = (hsb < 49 ? 0 : (hsb < 53 ? 1 : 2)) * 3 +
                    (wsb < 49 ? 0 : (wsb < 53 ? 1 : 2));
          int ridx = (ia - ib + 6) * 13 + (ja - jb + 6);
          float bias = rpb[ridx * kNH + head];
          sv = S[mt][nt][r] * scale + bias + ((rga != rgb) ? -100.f : 0.f);
        }
        vals[nt] = sv;
      }
      float mx = fmaxf(fmaxf(vals[0], vals[1]), fmaxf(vals[2], vals[3]));
      mx = row_max16(mx);
      float e[4], ss = 0.f;
      for (int nt = 0; nt < 4; ++nt) { e[nt] = __expf(vals[nt] - mx); ss += e[nt]; }
      ss = row_sum16(ss);
      float inv = 1.f / ss;
      for (int nt = 0; nt < 4; ++nt)
        Pm[m * 64 + nt * 16 + nloc] = (_Float16)(e[nt] * inv);
    }
  }

  // O = P @ V  (64x32, K = 64 in two steps); store rows < 49
  for (int nt2 = 0; nt2 < 2; ++nt2) {
    h16 b0 = frag_ld(vT, 64, nt2 * 16 + nloc, 0, lane);
    h16 b1 = frag_ld(vT, 64, nt2 * 16 + nloc, 32, lane);
    for (int mt = 0; mt < 4; ++mt) {
      h16 a0 = frag_ld(Pm, 64, mt * 16 + nloc, 0, lane);
      h16 a1 = frag_ld(Pm, 64, mt * 16 + nloc, 32, lane);
      f8 acc = {};
      acc = WMMA_F16(a0, b0, acc);
      acc = WMMA_F16(a1, b1, acc);
      int col = head * 32 + nt2 * 16 + nloc;
      int rb = mt * 16 + hi8;
      for (int r = 0; r < 8; ++r) {
        int m = rb + r;
        if (m < kN)
          out[((size_t)bw * kN + m) * kC + col] = (_Float16)acc[r];
      }
    }
  }
}

// ---------------------------------------------------------------------------
// Proj GEMM + bias + window reverse + unshift + residual -> x_new (f32, d_out)
__global__ void __launch_bounds__(256)
k_proj(const _Float16* __restrict__ ao, const _Float16* __restrict__ wp,
       const float* __restrict__ pb, const float* __restrict__ x,
       float* __restrict__ xo) {
  int wave = threadIdx.x >> 5, lane = threadIdx.x & 31;
  int bw = blockIdx.x;
  int b = bw >> 6, win = bw & 63, wh = win >> 3, ww = win & 7;
  int hi8 = (lane & 16) ? 8 : 0;
  const _Float16* A = ao + (size_t)bw * kN * kC;
  for (int t = wave; t < 48; t += 8) {            // 4 mtiles x 12 ntiles
    int mt = t & 3, nt = t >> 2;
    f8 acc = {};
    for (int kk = 0; kk < 6; ++kk) {
      h16 a = frag_ld(A, kC, mt * 16 + (lane & 15), kk * 32, lane);
      h16 bf = frag_ld(wp, kC, nt * 16 + (lane & 15), kk * 32, lane);
      acc = WMMA_F16(a, bf, acc);
    }
    int col = nt * 16 + (lane & 15);
    float bias = pb[col];
    int rb = mt * 16 + hi8;
    for (int r = 0; r < 8; ++r) {
      int m = rb + r;
      if (m < kN) {
        int i = m / 7, j = m - i * 7;
        int h = wh * 7 + i + 3; if (h >= 56) h -= 56;
        int w = ww * 7 + j + 3; if (w >= 56) w -= 56;
        size_t idx = ((size_t)b * kL + h * 56 + w) * kC + col;
        xo[idx] = x[idx] + acc[r] + bias;
      }
    }
  }
}

// ---------------------------------------------------------------------------
// FC1: (64 rows x 192) @ (192 x 768) + bias -> exact GELU -> f16
__global__ void __launch_bounds__(256)
k_fc1(const _Float16* __restrict__ xln, const _Float16* __restrict__ w1,
      const float* __restrict__ b1, _Float16* __restrict__ fo) {
  int wave = threadIdx.x >> 5, lane = threadIdx.x & 31;
  int row0 = blockIdx.x * 64;
  int hi8 = (lane & 16) ? 8 : 0;
  const _Float16* A = xln + (size_t)row0 * kC;
  for (int t = wave; t < 192; t += 8) {           // 4 mtiles x 48 ntiles
    int mt = t & 3, nt = t >> 2;
    f8 acc = {};
    for (int kk = 0; kk < 6; ++kk) {
      h16 a = frag_ld(A, kC, mt * 16 + (lane & 15), kk * 32, lane);
      h16 bf = frag_ld(w1, kC, nt * 16 + (lane & 15), kk * 32, lane);
      acc = WMMA_F16(a, bf, acc);
    }
    int col = nt * 16 + (lane & 15);
    float bias = b1[col];
    int rb = row0 + mt * 16 + hi8;
    for (int r = 0; r < 8; ++r) {
      int m = rb + r;
      float vv = acc[r] + bias;
      float gel = 0.5f * vv * (1.f + erff(vv * 0.70710678118654752f));
      fo[(size_t)m * kHID + col] = (_Float16)gel;
    }
  }
}

// ---------------------------------------------------------------------------
// FC2: (64 rows x 768) @ (768 x 192) + bias, accumulate into d_out
__global__ void __launch_bounds__(256)
k_fc2(const _Float16* __restrict__ fo, const _Float16* __restrict__ w2,
      const float* __restrict__ b2, float* __restrict__ xo) {
  int wave = threadIdx.x >> 5, lane = threadIdx.x & 31;
  int row0 = blockIdx.x * 64;
  int hi8 = (lane & 16) ? 8 : 0;
  const _Float16* A = fo + (size_t)row0 * kHID;
  for (int t = wave; t < 48; t += 8) {            // 4 mtiles x 12 ntiles
    int mt = t & 3, nt = t >> 2;
    f8 acc = {};
    for (int kk = 0; kk < 24; ++kk) {
      h16 a = frag_ld(A, kHID, mt * 16 + (lane & 15), kk * 32, lane);
      h16 bf = frag_ld(w2, kHID, nt * 16 + (lane & 15), kk * 32, lane);
      acc = WMMA_F16(a, bf, acc);
    }
    int col = nt * 16 + (lane & 15);
    float bias = b2[col];
    int rb = row0 + mt * 16 + hi8;
    for (int r = 0; r < 8; ++r) {
      int m = rb + r;
      size_t idx = (size_t)m * kC + col;
      xo[idx] += acc[r] + bias;
    }
  }
}

// ---------------------------------------------------------------------------
extern "C" void kernel_launch(void* const* d_in, const int* in_sizes, int n_in,
                              void* d_out, int out_size, void* d_ws, size_t ws_size,
                              hipStream_t stream) {
  (void)in_sizes; (void)n_in; (void)out_size; (void)ws_size;
  const float* x      = (const float*)d_in[0];
  const float* qkv_w  = (const float*)d_in[1];
  const float* qkv_b  = (const float*)d_in[2];
  const float* proj_w = (const float*)d_in[3];
  const float* proj_b = (const float*)d_in[4];
  const float* rpb    = (const float*)d_in[5];
  const float* n1g    = (const float*)d_in[6];
  const float* n1b    = (const float*)d_in[7];
  const float* n2g    = (const float*)d_in[8];
  const float* n2b    = (const float*)d_in[9];
  const float* w1     = (const float*)d_in[10];
  const float* b1     = (const float*)d_in[11];
  const float* w2     = (const float*)d_in[12];
  const float* b2     = (const float*)d_in[13];
  float* out = (float*)d_out;
  char*  ws  = (char*)d_ws;

  const size_t MB = 1024ull * 1024ull;
  _Float16* wqkv_h  = (_Float16*)(ws + 0);           // 576*192
  _Float16* wproj_h = (_Float16*)(ws + 221440);      // 192*192
  _Float16* w1_h    = (_Float16*)(ws + 295424);      // 768*192
  _Float16* w2_h    = (_Float16*)(ws + 590592);      // 192*768
  _Float16* winA    = (_Float16*)(ws + 1 * MB);      // 2048*49*192   (later: attn out)
  _Float16* qbuf    = (_Float16*)(ws + 40 * MB);     // 2048*6*49*32
  _Float16* kbuf    = (_Float16*)(ws + 80 * MB);
  _Float16* vbuf    = (_Float16*)(ws + 120 * MB);
  _Float16* xln     = (_Float16*)(ws + 160 * MB);    // 100352*192
  _Float16* attn_o  = winA;                          // reuse (winA dead after QKV)
  _Float16* fc1     = winA;                          // reuse win+qkv region (dead)

  // weights -> f16
  k_f2h<<<(3 * kC * kC + 255) / 256, 256, 0, stream>>>(qkv_w, wqkv_h, 3 * kC * kC);
  k_f2h<<<(kC * kC + 255) / 256, 256, 0, stream>>>(proj_w, wproj_h, kC * kC);
  k_f2h<<<(kHID * kC + 255) / 256, 256, 0, stream>>>(w1, w1_h, kHID * kC);
  k_f2h<<<(kC * kHID + 255) / 256, 256, 0, stream>>>(w2, w2_h, kC * kHID);

  // LN1 + shift + window partition
  k_ln1_win<<<kTOK / 8, 256, 0, stream>>>(x, n1g, n1b, winA);
  // QKV projection
  k_qkv<<<kBW, 256, 0, stream>>>(winA, wqkv_h, qkv_b, qbuf, kbuf, vbuf);
  // windowed attention (one wave per window*head)
  k_attn<<<(kBW * kNH) / 4, 128, 0, stream>>>(qbuf, kbuf, vbuf, rpb, attn_o);
  // output projection + reverse + residual -> d_out (x_new)
  k_proj<<<kBW, 256, 0, stream>>>(attn_o, wproj_h, proj_b, x, out);
  // LN2
  k_ln2<<<kTOK / 8, 256, 0, stream>>>(out, n2g, n2b, xln);
  // MLP
  k_fc1<<<kTOK / 64, 256, 0, stream>>>(xln, w1_h, b1, fc1);
  k_fc2<<<kTOK / 64, 256, 0, stream>>>(fc1, w2_h, b2, out);
}
